// SequenceParallelAttention_70068096467925
// MI455X (gfx1250) — compile-verified, matrix operand order checked by reading, not run on hardware
//
#include <hip/hip_runtime.h>

// ---------------------------------------------------------------------------
// MHA block for MI455X (gfx1250): bf16 WMMA everywhere, fp32 accumulate.
//   y = softmax((xWq+bq)(xWk+bk)^T / sqrt(d)) (xWv+bv) Wout^T + bout
// ---------------------------------------------------------------------------

typedef __attribute__((ext_vector_type(16))) __bf16         v16bf;
typedef __attribute__((ext_vector_type(8)))  float          v8f;
typedef __attribute__((ext_vector_type(4)))  unsigned int   v4u;
typedef __attribute__((ext_vector_type(4)))  float          v4f;
typedef __attribute__((ext_vector_type(4)))  unsigned short v4s;

#define HIDDEN 2048
#define SEQ    2048
#define HEADS  16
#define HDIM   128
#define BATCH  2

// gfx1250 async global->LDS copy path (ASYNCcnt-tracked), with fallback.
#if defined(__AMDGCN__) && __has_builtin(__builtin_amdgcn_global_load_async_to_lds_b128)
#define HAVE_ASYNC_LDS 1
typedef __attribute__((ext_vector_type(4))) int v4i;
typedef __attribute__((address_space(1))) v4i* as1_v4i;  // global src
typedef __attribute__((address_space(3))) v4i* as3_v4i;  // LDS dst
#endif

__device__ __forceinline__ unsigned short f2bf(float x) {
  unsigned u = __float_as_uint(x);
  u += 0x7FFFu + ((u >> 16) & 1u);          // round-to-nearest-even
  return (unsigned short)(u >> 16);
}

__device__ __forceinline__ v8f vzero8() {
  v8f z;
#pragma unroll
  for (int i = 0; i < 8; i++) z[i] = 0.0f;
  return z;
}

union FragU {
  v16bf v;
  v4u   q[2];
};

// Load one 16x32 bf16 WMMA operand fragment. Caller pre-offsets the pointer
// by the per-lane row and the (lane>=16 ? +8 : +0) K-select. Two b128 loads:
// K[+0..7] and K[+16..23] for lanes 0-15; K[+8..15] / K[+24..31] for 16-31.
__device__ __forceinline__ v16bf load_frag(const unsigned short* p) {
  FragU f;
  f.q[0] = *reinterpret_cast<const v4u*>(p);
  f.q[1] = *reinterpret_cast<const v4u*>(p + 16);
  return f.v;
}

// Copy 16 bytes global -> LDS (async when available).
__device__ __forceinline__ void cp_g2s_b128(const unsigned short* g,
                                            unsigned short* l) {
#ifdef HAVE_ASYNC_LDS
  __builtin_amdgcn_global_load_async_to_lds_b128(
      (as1_v4i)(void*)g, (as3_v4i)(void*)l, 0, 0);
#else
  *reinterpret_cast<v4u*>(l) = *reinterpret_cast<const v4u*>(g);
#endif
}

__device__ __forceinline__ void wait_async_copies() {
#ifdef HAVE_ASYNC_LDS
#if __has_builtin(__builtin_amdgcn_s_wait_asynccnt)
  __builtin_amdgcn_s_wait_asynccnt(0);
#else
  asm volatile("s_wait_asynccnt 0" ::: "memory");
#endif
#endif
}

// ---------------------------------------------------------------------------
// fp32 -> bf16 conversion (vectorized 4 elems/thread)
// ---------------------------------------------------------------------------
__global__ void cvt_f32_bf16(const float* __restrict__ in,
                             unsigned short* __restrict__ out, int n4) {
  int i = blockIdx.x * blockDim.x + threadIdx.x;
  if (i < n4) {
    v4f f = reinterpret_cast<const v4f*>(in)[i];
    v4s o;
#pragma unroll
    for (int j = 0; j < 4; j++) o[j] = f2bf(f[j]);
    reinterpret_cast<v4s*>(out)[i] = o;
  }
}

// ---------------------------------------------------------------------------
// NT GEMM:  C[M,N] = A[M,K] * B[N,K]^T + bias[N]
//   MODE 0: bf16 row-major output               (Q, K projections)
//   MODE 1: bf16 transposed-per-batch output V^T [b, HIDDEN, SEQ]  (V proj)
//   MODE 2: fp32 row-major output               (final out projection)
// Block tile 128x128, 256 threads = 8 waves in a 4(M) x 2(N) grid,
// each wave owns 32x64 = 2x4 WMMA tiles.
// ---------------------------------------------------------------------------
template <int MODE>
__global__ __launch_bounds__(256) void gemm_nt_bf16(
    const unsigned short* __restrict__ A,  // [M,K] bf16
    const unsigned short* __restrict__ B,  // [N,K] bf16
    const float* __restrict__ bias,        // [N]
    void* __restrict__ C, int M, int N, int K) {
  const int lane  = threadIdx.x & 31;
  const int wave  = threadIdx.x >> 5;
  const int lrow  = lane & 15;
  const int sel   = (lane & 16) ? 8 : 0;
  const int rbase = (lane & 16) ? 8 : 0;

  const int m0 = blockIdx.y * 128 + (wave >> 1) * 32;
  const int n0 = blockIdx.x * 128 + (wave & 1) * 64;

  v8f acc[2][4];
#pragma unroll
  for (int i = 0; i < 2; i++)
#pragma unroll
    for (int j = 0; j < 4; j++) acc[i][j] = vzero8();

  const unsigned short* Abase = A + (size_t)(m0 + lrow) * K;
  const unsigned short* Bbase = B + (size_t)(n0 + lrow) * K;

  for (int k0 = 0; k0 < K; k0 += 32) {
    // Pull the streams ~8 K-steps ahead into the near cache.
    if (k0 + 256 < K) {
      __builtin_prefetch(Abase + k0 + 256 + sel, 0, 3);
      __builtin_prefetch(Bbase + k0 + 256 + sel, 0, 3);
    }
    v16bf af[2], bfr[4];
#pragma unroll
    for (int i = 0; i < 2; i++)
      af[i] = load_frag(Abase + (size_t)i * 16 * K + k0 + sel);
#pragma unroll
    for (int j = 0; j < 4; j++)
      bfr[j] = load_frag(Bbase + (size_t)j * 16 * K + k0 + sel);
#pragma unroll
    for (int i = 0; i < 2; i++)
#pragma unroll
      for (int j = 0; j < 4; j++)
        acc[i][j] = __builtin_amdgcn_wmma_f32_16x16x32_bf16(
            false, af[i], false, bfr[j], (short)0, acc[i][j], false, false);
  }

  float bv[4];
#pragma unroll
  for (int j = 0; j < 4; j++) bv[j] = bias[n0 + j * 16 + lrow];

#pragma unroll
  for (int i = 0; i < 2; i++) {
#pragma unroll
    for (int j = 0; j < 4; j++) {
      const int nn = n0 + j * 16 + lrow;
      if (MODE == 1) {
        // C-fragment rows are 8 consecutive seq positions per lane: pack and
        // store V^T[b][nn][s0..s0+7] with two 8-byte writes.
        v4s pk0, pk1;
#pragma unroll
        for (int r = 0; r < 4; r++) pk0[r] = f2bf(acc[i][j][r] + bv[j]);
#pragma unroll
        for (int r = 0; r < 4; r++) pk1[r] = f2bf(acc[i][j][r + 4] + bv[j]);
        const int mrow = m0 + i * 16 + rbase;  // first packed global M index
        const int bb = mrow >> 11;             // batch = mrow / SEQ
        const int ss = mrow & (SEQ - 1);
        unsigned short* dst =
            (unsigned short*)C + ((size_t)bb * HIDDEN + nn) * SEQ + ss;
        *reinterpret_cast<v4s*>(dst) = pk0;
        *reinterpret_cast<v4s*>(dst + 4) = pk1;
      } else {
#pragma unroll
        for (int r = 0; r < 8; r++) {
          const int mm = m0 + i * 16 + rbase + r;
          const float val = acc[i][j][r] + bv[j];
          if (MODE == 2)
            ((float*)C)[(size_t)mm * N + nn] = val;
          else
            ((unsigned short*)C)[(size_t)mm * N + nn] = f2bf(val);
        }
      }
    }
  }
}

// ---------------------------------------------------------------------------
// Flash-style attention with double-buffered async LDS staging.
// One workgroup = one (b,h) x 128 query rows; 8 waves, each owning 16 rows.
// While block i is consumed from LDS buffer (i&1), the async DMA for block
// i+1 streams into buffer (~i&1); ASYNCcnt is drained only at the end of the
// iteration, so the copy latency overlaps the 64 WMMAs of the compute phase.
// ---------------------------------------------------------------------------
__global__ __launch_bounds__(256) void attn_fwd(
    const unsigned short* __restrict__ Q,   // [B*S, HIDDEN] bf16
    const unsigned short* __restrict__ Km,  // [B*S, HIDDEN] bf16
    const unsigned short* __restrict__ Vt,  // [B, HIDDEN, S] bf16 (V^T)
    unsigned short* __restrict__ ctx) {     // [B*S, HIDDEN] bf16
  const int lane  = threadIdx.x & 31;
  const int wave  = threadIdx.x >> 5;
  const int lrow  = lane & 15;
  const int sel   = (lane & 16) ? 8 : 0;
  const int rbase = (lane & 16) ? 8 : 0;

  const int b  = blockIdx.x >> 4;
  const int h  = blockIdx.x & 15;
  const int q0 = blockIdx.y * 128 + wave * 16;

  __shared__ unsigned short kls[2][128 * 128];  // 2 x 32 KB K tiles
  __shared__ unsigned short vls[2][128 * 128];  // 2 x 32 KB V^T tiles
  __shared__ unsigned short pls[8][16 * 128];   // 32 KB per-wave P tiles
  unsigned short* myp = &pls[wave][0];

  const unsigned short* gkbase = Km + ((size_t)b * SEQ) * HIDDEN + h * HDIM;
  const unsigned short* gvbase = Vt + ((size_t)b * HIDDEN + h * HDIM) * SEQ;

  // Cooperative staging of one 128x128 K tile + 128x128 V^T tile (64 KB):
  // 16 async b128 instructions per thread.
  auto stage = [&](int kv0, int buf) {
    const unsigned short* gk = gkbase + (size_t)kv0 * HIDDEN;
    const unsigned short* gv = gvbase + kv0;
#pragma unroll
    for (int i = 0; i < 8; i++) {
      const int c   = threadIdx.x + 256 * i;  // 2048 16B-chunks per tile
      const int row = c >> 4;
      const int col = (c & 15) * 8;
      cp_g2s_b128(gk + (size_t)row * HIDDEN + col, &kls[buf][row * 128 + col]);
      cp_g2s_b128(gv + (size_t)row * SEQ + col, &vls[buf][row * 128 + col]);
    }
  };

  // Q fragments (d = 128 -> 4 K-steps), reused across all KV blocks.
  v16bf qf[4];
  const unsigned short* Qrow =
      Q + ((size_t)b * SEQ + q0 + lrow) * HIDDEN + h * HDIM;
#pragma unroll
  for (int ks = 0; ks < 4; ks++) qf[ks] = load_frag(Qrow + ks * 32 + sel);

  v8f oacc[8];
#pragma unroll
  for (int t = 0; t < 8; t++) oacc[t] = vzero8();
  float mrow[8], lsum[8];
#pragma unroll
  for (int r = 0; r < 8; r++) { mrow[r] = -1.0e30f; lsum[r] = 0.0f; }

  const float scale = 0.08838834764831845f;  // 1/sqrt(128)

  // Prologue: stage block 0 and wait for it.
  stage(0, 0);
  wait_async_copies();
  __syncthreads();

  int cur = 0;
  for (int kv0 = 0; kv0 < SEQ; kv0 += 128) {
    // Kick off DMA for the next block into the other buffer (overlapped).
    if (kv0 + 128 < SEQ) stage(kv0 + 128, cur ^ 1);

    // ---- S = Q K^T (16 x 128 per wave), K operand from LDS ----
    v8f sacc[8];
#pragma unroll
    for (int t = 0; t < 8; t++) sacc[t] = vzero8();
#pragma unroll
    for (int nt = 0; nt < 8; nt++) {
      const unsigned short* Krow = &kls[cur][(nt * 16 + lrow) * 128];
#pragma unroll
      for (int ks = 0; ks < 4; ks++) {
        v16bf kf = load_frag(Krow + ks * 32 + sel);
        sacc[nt] = __builtin_amdgcn_wmma_f32_16x16x32_bf16(
            false, qf[ks], false, kf, (short)0, sacc[nt], false, false);
      }
    }

    // ---- online softmax: 16-lane row reductions match C-frag row ownership
    float tmax[8];
#pragma unroll
    for (int r = 0; r < 8; r++) tmax[r] = -1.0e30f;
#pragma unroll
    for (int nt = 0; nt < 8; nt++)
#pragma unroll
      for (int r = 0; r < 8; r++) {
        float s = sacc[nt][r] * scale;
        sacc[nt][r] = s;
        tmax[r] = fmaxf(tmax[r], s);
      }
#pragma unroll
    for (int mask = 8; mask >= 1; mask >>= 1)
#pragma unroll
      for (int r = 0; r < 8; r++)
        tmax[r] = fmaxf(tmax[r], __shfl_xor(tmax[r], mask, 32));

    float alpha[8];
#pragma unroll
    for (int r = 0; r < 8; r++) {
      float mn = fmaxf(mrow[r], tmax[r]);
      alpha[r] = __expf(mrow[r] - mn);
      mrow[r] = mn;
    }
    float tsum[8];
#pragma unroll
    for (int r = 0; r < 8; r++) tsum[r] = 0.0f;
#pragma unroll
    for (int nt = 0; nt < 8; nt++)
#pragma unroll
      for (int r = 0; r < 8; r++) {
        float p = __expf(sacc[nt][r] - mrow[r]);
        sacc[nt][r] = p;
        tsum[r] += p;
      }
#pragma unroll
    for (int mask = 8; mask >= 1; mask >>= 1)
#pragma unroll
      for (int r = 0; r < 8; r++) tsum[r] += __shfl_xor(tsum[r], mask, 32);
#pragma unroll
    for (int r = 0; r < 8; r++) lsum[r] = lsum[r] * alpha[r] + tsum[r];
#pragma unroll
    for (int nt = 0; nt < 8; nt++)
#pragma unroll
      for (int r = 0; r < 8; r++) oacc[nt][r] *= alpha[r];

    // ---- P: C-frag -> LDS (row-major 16x128 bf16) -> A-frag ----
    // Per-wave private region; DS ops are in-order within a wave.
#pragma unroll
    for (int nt = 0; nt < 8; nt++)
#pragma unroll
      for (int r = 0; r < 8; r++)
        myp[(rbase + r) * 128 + nt * 16 + lrow] = f2bf(sacc[nt][r]);

    v16bf pf[4];
#pragma unroll
    for (int ks = 0; ks < 4; ks++)
      pf[ks] = load_frag(myp + lrow * 128 + ks * 32 + sel);

    // ---- O += P @ V, V^T operand from LDS ----
#pragma unroll
    for (int nt = 0; nt < 8; nt++) {
      const unsigned short* Vrow = &vls[cur][(nt * 16 + lrow) * 128];
#pragma unroll
      for (int ks = 0; ks < 4; ks++) {
        v16bf vf = load_frag(Vrow + ks * 32 + sel);
        oacc[nt] = __builtin_amdgcn_wmma_f32_16x16x32_bf16(
            false, pf[ks], false, vf, (short)0, oacc[nt], false, false);
      }
    }

    // Drain this wave's outstanding async copies (next tile) and rendezvous:
    // after the barrier the next buffer is valid for every wave, and this
    // buffer may be overwritten by the following stage().
    wait_async_copies();
    __syncthreads();
    cur ^= 1;
  }

  // ---- epilogue: O /= l, write context [b, s, h*128 + d] ----
#pragma unroll
  for (int r = 0; r < 8; r++) lsum[r] = 1.0f / lsum[r];
#pragma unroll
  for (int nt = 0; nt < 8; nt++) {
    const int col = h * HDIM + nt * 16 + lrow;
#pragma unroll
    for (int r = 0; r < 8; r++) {
      const int srow = q0 + rbase + r;
      ctx[((size_t)b * SEQ + srow) * HIDDEN + col] = f2bf(oacc[nt][r] * lsum[r]);
    }
  }
}

// ---------------------------------------------------------------------------
extern "C" void kernel_launch(void* const* d_in, const int* in_sizes, int n_in,
                              void* d_out, int out_size, void* d_ws,
                              size_t ws_size, hipStream_t stream) {
  (void)in_sizes; (void)n_in; (void)out_size; (void)ws_size;

  const float* x  = (const float*)d_in[0];
  const float* Wq = (const float*)d_in[1];
  const float* bq = (const float*)d_in[2];
  const float* Wk = (const float*)d_in[3];
  const float* bk = (const float*)d_in[4];
  const float* Wv = (const float*)d_in[5];
  const float* bv = (const float*)d_in[6];
  const float* Wo = (const float*)d_in[7];
  const float* bo = (const float*)d_in[8];

  char* ws = (char*)d_ws;
  const size_t MB = 1024ull * 1024ull;
  unsigned short* xb  = (unsigned short*)(ws + 0 * MB);   // 16 MB
  unsigned short* wqb = (unsigned short*)(ws + 16 * MB);  // 8 MB each
  unsigned short* wkb = (unsigned short*)(ws + 24 * MB);
  unsigned short* wvb = (unsigned short*)(ws + 32 * MB);
  unsigned short* wob = (unsigned short*)(ws + 40 * MB);
  unsigned short* Qb  = (unsigned short*)(ws + 48 * MB);  // 16 MB each
  unsigned short* Kb  = (unsigned short*)(ws + 64 * MB);
  unsigned short* Vtb = (unsigned short*)(ws + 80 * MB);
  unsigned short* ctx = (unsigned short*)(ws + 96 * MB);

  const int M = BATCH * SEQ, N = HIDDEN, K = HIDDEN;

  cvt_f32_bf16<<<(M * K / 4 + 255) / 256, 256, 0, stream>>>(x, xb, M * K / 4);
  cvt_f32_bf16<<<(N * K / 4 + 255) / 256, 256, 0, stream>>>(Wq, wqb, N * K / 4);
  cvt_f32_bf16<<<(N * K / 4 + 255) / 256, 256, 0, stream>>>(Wk, wkb, N * K / 4);
  cvt_f32_bf16<<<(N * K / 4 + 255) / 256, 256, 0, stream>>>(Wv, wvb, N * K / 4);
  cvt_f32_bf16<<<(N * K / 4 + 255) / 256, 256, 0, stream>>>(Wo, wob, N * K / 4);

  dim3 gg(N / 128, M / 128), bb(256);
  gemm_nt_bf16<0><<<gg, bb, 0, stream>>>(xb, wqb, bq, Qb, M, N, K);
  gemm_nt_bf16<0><<<gg, bb, 0, stream>>>(xb, wkb, bk, Kb, M, N, K);
  gemm_nt_bf16<1><<<gg, bb, 0, stream>>>(xb, wvb, bv, Vtb, M, N, K);

  attn_fwd<<<dim3(BATCH * HEADS, SEQ / 128), 256, 0, stream>>>(Qb, Kb, Vtb, ctx);

  gemm_nt_bf16<2><<<gg, bb, 0, stream>>>(ctx, wob, bo, d_out, M, N, K);
}
